// video_transformer_3745211482877
// MI455X (gfx1250) — matrix operand order repaired and checked by reading, not compiled, not run on hardware
//
#include <hip/hip_runtime.h>
#include <hip/hip_bf16.h>

// ---------------------------------------------------------------------------
// Video-transformer cross-attention block, algebraically refactored:
//   logits = (q @ k_w) . LN(x)         (K projection eliminated)
//   attn   = (softmax @ LN(x)) @ v_w.T (V projection shrunk 128x)
// Attention core: SINGLE streaming pass over video (268 MB -> ~11.5us at
// 23.3 TB/s) using online softmax with wave-private state (no barriers in the
// hot loop). GEMMs use v_wmma_f32_16x16x32_bf16 with hi/lo bf16 split.
// ---------------------------------------------------------------------------

#define BATCH 512
#define FRAMES 128
#define DIM 1024
#define KT_TILES (DIM / 32)   // 32 K-tiles of 32
#define LN_EPS 1e-5f

typedef float v8f   __attribute__((ext_vector_type(8)));
typedef __bf16 v16bf __attribute__((ext_vector_type(16)));
typedef __bf16 v8bf  __attribute__((ext_vector_type(8)));

// ---------------- bf16 helpers (bit ops only; RNE rounding) ----------------
__device__ __forceinline__ __bf16 f32_to_bf16(float x) {
    union { float f; unsigned u; } c; c.f = x;
    unsigned r = c.u + 0x7FFFu + ((c.u >> 16) & 1u);
    union { unsigned short s; __bf16 h; } o; o.s = (unsigned short)(r >> 16);
    return o.h;
}
__device__ __forceinline__ float bf16_to_f32(__bf16 h) {
    union { unsigned short s; __bf16 h; } i; i.h = h;
    union { unsigned u; float f; } o; o.u = ((unsigned)i.s) << 16;
    return o.f;
}
__device__ __forceinline__ void split_hi_lo(float v, __bf16* hi, __bf16* lo) {
    __bf16 h = f32_to_bf16(v);
    *hi = h;
    *lo = f32_to_bf16(v - bf16_to_f32(h));
}

__device__ __forceinline__ v8f wmma_bf16(v16bf a, v16bf b, v8f c) {
    return __builtin_amdgcn_wmma_f32_16x16x32_bf16(false, a, false, b,
                                                   (short)0, c, false, false);
}

// ---------------- reductions (wave32) --------------------------------------
// Butterfly: result broadcast to all 32 lanes, no LDS, no barrier.
__device__ __forceinline__ float wave_bsum(float v) {
    for (int o = 16; o > 0; o >>= 1) v += __shfl_xor(v, o);
    return v;
}
__device__ __forceinline__ float warp_sum(float v) {
    for (int o = 16; o > 0; o >>= 1) v += __shfl_down(v, o);
    return v;
}
// Block-wide reduce of 2 values (used by one-shot LN kernels only).
__device__ __forceinline__ void block_sum2(float& a, float& b, float* s) {
    int lane = threadIdx.x & 31, wid = threadIdx.x >> 5;
    a = warp_sum(a); b = warp_sum(b);
    if (lane == 0) { s[wid] = a; s[8 + wid] = b; }
    __syncthreads();
    if (wid == 0) {
        float x = (lane < 8) ? s[lane] : 0.f;
        float y = (lane < 8) ? s[8 + lane] : 0.f;
        x = warp_sum(x); y = warp_sum(y);
        if (lane == 0) { s[0] = x; s[8] = y; }
    }
    __syncthreads();
    a = s[0]; b = s[8];
    __syncthreads();
}

// ---------------------------------------------------------------------------
// Kernel 1: pack a DxD fp32 weight into bf16 hi/lo WMMA B-fragments.
//  B[k][n] = transpose ? W[n][k] : W[k][n]
//  idx = (((ntile*KT + ktile)*32 + lane)*16 + j)
//  lane<16: n = lane,     K = ktile*32 + {0..7,16..23}
//  lane>=16: n = lane-16, K = ktile*32 + {8..15,24..31}
// ---------------------------------------------------------------------------
__global__ void vt_prep_weights(const float* __restrict__ W,
                                __bf16* __restrict__ Hi, __bf16* __restrict__ Lo,
                                int transpose) {
    int id = blockIdx.x * 256 + threadIdx.x;      // 0 .. 2^20-1
    int j    = id & 15;
    int lane = (id >> 4) & 31;
    int kt   = (id >> 9) & 31;
    int nt   = id >> 14;
    int n = nt * 16 + (lane & 15);
    int k = kt * 32 + ((j < 8) ? j : (8 + j)) + ((lane < 16) ? 0 : 8);
    float v = transpose ? W[(size_t)n * DIM + k] : W[(size_t)k * DIM + n];
    split_hi_lo(v, &Hi[id], &Lo[id]);
}

// ---------------------------------------------------------------------------
// Kernel 2: row LayerNorm; optional second input (residual add) before LN.
// ---------------------------------------------------------------------------
__global__ void vt_ln_rows(const float* __restrict__ X, const float* __restrict__ X2,
                           const float* __restrict__ g, const float* __restrict__ bb,
                           float* __restrict__ Y) {
    __shared__ float red[16];
    int row = blockIdx.x, t = threadIdx.x, e0 = t * 4;
    const float* xr = X + (size_t)row * DIM;
    float4 x = *reinterpret_cast<const float4*>(xr + e0);
    if (X2) {
        float4 y = *reinterpret_cast<const float4*>(X2 + (size_t)row * DIM + e0);
        x.x += y.x; x.y += y.y; x.z += y.z; x.w += y.w;
    }
    float sx  = x.x + x.y + x.z + x.w;
    float sxx = x.x * x.x + x.y * x.y + x.z * x.z + x.w * x.w;
    block_sum2(sx, sxx, red);
    float mu   = sx * (1.f / DIM);
    float var  = sxx * (1.f / DIM) - mu * mu;
    float rstd = rsqrtf(var + LN_EPS);
    float4 g4 = *reinterpret_cast<const float4*>(g + e0);
    float4 b4 = *reinterpret_cast<const float4*>(bb + e0);
    float4 o;
    o.x = (x.x - mu) * rstd * g4.x + b4.x;
    o.y = (x.y - mu) * rstd * g4.y + b4.y;
    o.z = (x.z - mu) * rstd * g4.z + b4.z;
    o.w = (x.w - mu) * rstd * g4.w + b4.w;
    *reinterpret_cast<float4*>(Y + (size_t)row * DIM + e0) = o;
}

// ---------------------------------------------------------------------------
// Kernel 3: GEMM  Y[M,1024] = X[M,1024] @ Bpacked + bias   (fp32 in/out)
// BM=64 BN=64 BK=32; 8 waves (4m x 2n); hi/lo bf16 split, 6 WMMA per K-step.
// ---------------------------------------------------------------------------
__global__ void vt_gemm_bias(const float* __restrict__ X,
                             const __bf16* __restrict__ WH,
                             const __bf16* __restrict__ WL,
                             const float* __restrict__ bias,
                             float* __restrict__ Y) {
    __shared__ __align__(16) __bf16 Ah[64 * 40];   // padded stride 40
    __shared__ __align__(16) __bf16 Al[64 * 40];

    const int t = threadIdx.x;
    const int lane = t & 31, wv = t >> 5;
    const int wm = wv >> 1, wn = wv & 1;
    const int m0 = blockIdx.y * 64, n0 = blockIdx.x * 64;

    v8f acc0 = {0.f, 0.f, 0.f, 0.f, 0.f, 0.f, 0.f, 0.f};
    v8f acc1 = {0.f, 0.f, 0.f, 0.f, 0.f, 0.f, 0.f, 0.f};

    const int arow = wm * 16 + (lane & 15);
    const int abase = (lane < 16) ? 0 : 8;          // A 16-bit fragment K offset
    const int sr = t >> 3;                           // stage row 0..31
    const int sc = (t & 7) * 4;                      // stage col
    const int ntg0 = blockIdx.x * 4 + wn * 2;        // global n-tile for nt=0
    const v16bf* BH = reinterpret_cast<const v16bf*>(WH);
    const v16bf* BL = reinterpret_cast<const v16bf*>(WL);

    for (int kt = 0; kt < KT_TILES; ++kt) {
        const int gk = kt * 32 + sc;
#pragma unroll
        for (int rr = 0; rr < 2; ++rr) {
            int row = sr + rr * 32;
            float4 x4 = *reinterpret_cast<const float4*>(
                X + (size_t)(m0 + row) * DIM + gk);
            split_hi_lo(x4.x, &Ah[row * 40 + sc + 0], &Al[row * 40 + sc + 0]);
            split_hi_lo(x4.y, &Ah[row * 40 + sc + 1], &Al[row * 40 + sc + 1]);
            split_hi_lo(x4.z, &Ah[row * 40 + sc + 2], &Al[row * 40 + sc + 2]);
            split_hi_lo(x4.w, &Ah[row * 40 + sc + 3], &Al[row * 40 + sc + 3]);
        }
        __syncthreads();

        // A fragments: lane<16 -> K {0..7,16..23}; lane>=16 -> {8..15,24..31}
        const __bf16* arh = &Ah[arow * 40];
        v8bf ph0 = *reinterpret_cast<const v8bf*>(arh + abase);
        v8bf ph1 = *reinterpret_cast<const v8bf*>(arh + abase + 16);
        v16bf afh = __builtin_shufflevector(ph0, ph1, 0, 1, 2, 3, 4, 5, 6, 7,
                                            8, 9, 10, 11, 12, 13, 14, 15);
        const __bf16* arl = &Al[arow * 40];
        v8bf pl0 = *reinterpret_cast<const v8bf*>(arl + abase);
        v8bf pl1 = *reinterpret_cast<const v8bf*>(arl + abase + 16);
        v16bf afl = __builtin_shufflevector(pl0, pl1, 0, 1, 2, 3, 4, 5, 6, 7,
                                            8, 9, 10, 11, 12, 13, 14, 15);

        size_t bi0 = ((size_t)ntg0 * KT_TILES + kt) * 32 + lane;
        size_t bi1 = ((size_t)(ntg0 + 1) * KT_TILES + kt) * 32 + lane;
        v16bf bh0 = BH[bi0], bl0 = BL[bi0];
        v16bf bh1 = BH[bi1], bl1 = BL[bi1];

        acc0 = wmma_bf16(afh, bh0, acc0);
        acc0 = wmma_bf16(afl, bh0, acc0);
        acc0 = wmma_bf16(afh, bl0, acc0);
        acc1 = wmma_bf16(afh, bh1, acc1);
        acc1 = wmma_bf16(afl, bh1, acc1);
        acc1 = wmma_bf16(afh, bl1, acc1);
        __syncthreads();
    }

    const int mrow = m0 + wm * 16 + ((lane & 16) ? 8 : 0);
#pragma unroll
    for (int nt = 0; nt < 2; ++nt) {
        int gn = n0 + wn * 32 + nt * 16 + (lane & 15);
        float bv = bias ? bias[gn] : 0.f;
        v8f a = nt ? acc1 : acc0;
#pragma unroll
        for (int r = 0; r < 8; ++r)
            Y[(size_t)(mrow + r) * DIM + gn] = a[r] + bv;
    }
}

// ---------------------------------------------------------------------------
// Kernel 4: fused attention core, SINGLE pass over video.
// Block = one batch row b (256 threads = 8 waves). Wave w owns frames
// [w*16, w*16+16); each lane holds a 32-element register slice of the
// 1024-dim row (e = j*128 + lane*4, j=0..7 -> coalesced 512B per load).
// Per frame: wave-local butterfly reductions (zero barriers) give LN stats
// and the folded-LN logit; online softmax keeps (m, s, acc[32], cacc) in
// registers. Final 8-way cross-wave flash combine through 32KB LDS.
// ---------------------------------------------------------------------------
__global__ void vt_attn_fused(const float* __restrict__ video,
                              const float* __restrict__ QK,
                              const float* __restrict__ Q,
                              const float* __restrict__ k_b,
                              const float* __restrict__ g,
                              const float* __restrict__ bln,
                              float* __restrict__ XW) {
    __shared__ __align__(16) float accL[8 * DIM];  // 32 KB
    __shared__ float wmx[8], wsm[8], wcc[8];

    const int t = threadIdx.x, lane = t & 31, w = t >> 5;
    const int b = blockIdx.x;
    const float* qk = QK + (size_t)b * DIM;
    const float* q  = Q  + (size_t)b * DIM;

    // Preload qg = qk*g into registers; wave-local sums of the frame-invariant
    // scalars (every wave covers the full 1024 dims -> no block reduce).
    float4 qg[8];
    float c1 = 0.f, c2 = 0.f, c3 = 0.f;
#pragma unroll
    for (int j = 0; j < 8; ++j) {
        int e = j * 128 + lane * 4;
        float4 qk4 = *reinterpret_cast<const float4*>(qk + e);
        float4 g4  = *reinterpret_cast<const float4*>(g + e);
        float4 b4  = *reinterpret_cast<const float4*>(bln + e);
        float4 q4  = *reinterpret_cast<const float4*>(q + e);
        float4 kb4 = *reinterpret_cast<const float4*>(k_b + e);
        qg[j].x = qk4.x * g4.x; qg[j].y = qk4.y * g4.y;
        qg[j].z = qk4.z * g4.z; qg[j].w = qk4.w * g4.w;
        c1 += qg[j].x + qg[j].y + qg[j].z + qg[j].w;
        c2 += qk4.x * b4.x + qk4.y * b4.y + qk4.z * b4.z + qk4.w * b4.w;
        c3 += q4.x * kb4.x + q4.y * kb4.y + q4.z * kb4.z + q4.w * kb4.w;
    }
    c1 = wave_bsum(c1); c2 = wave_bsum(c2); c3 = wave_bsum(c3);

    // Online softmax state (wave-private, frames w*16 .. w*16+15).
    float m = -3.4e38f, s = 0.f, cacc = 0.f;
    float4 acc[8];
#pragma unroll
    for (int j = 0; j < 8; ++j) acc[j] = make_float4(0.f, 0.f, 0.f, 0.f);

    const float* vbase = video + ((size_t)b * FRAMES + w * 16) * DIM;
    for (int ff = 0; ff < 16; ++ff) {
        const float* vr = vbase + (size_t)ff * DIM;
        if (ff + 1 < 16)
            __builtin_prefetch(vr + DIM + lane * 4, 0, 0);
        float4 x[8];
        float sx = 0.f, sxx = 0.f, sq = 0.f;
#pragma unroll
        for (int j = 0; j < 8; ++j) {
            x[j] = *reinterpret_cast<const float4*>(vr + j * 128 + lane * 4);
            sx  += x[j].x + x[j].y + x[j].z + x[j].w;
            sxx += x[j].x * x[j].x + x[j].y * x[j].y
                 + x[j].z * x[j].z + x[j].w * x[j].w;
            sq  += qg[j].x * x[j].x + qg[j].y * x[j].y
                 + qg[j].z * x[j].z + qg[j].w * x[j].w;
        }
        sx = wave_bsum(sx); sxx = wave_bsum(sxx); sq = wave_bsum(sq);

        float mu   = sx * (1.f / DIM);
        float var  = sxx * (1.f / DIM) - mu * mu;
        float rstd = rsqrtf(var + LN_EPS);
        float l = (rstd * (sq - mu * c1) + c2 + c3) * 0.03125f;  // /sqrt(1024)

        float mn = fmaxf(m, l);
        float sc = __expf(m - mn);      // 0 on first iter (m = -inf)
        float p  = __expf(l - mn);
        s    = s * sc + p;
        cacc = cacc * sc + p * rstd * mu;
        float af = p * rstd;
#pragma unroll
        for (int j = 0; j < 8; ++j) {
            acc[j].x = acc[j].x * sc + af * x[j].x;
            acc[j].y = acc[j].y * sc + af * x[j].y;
            acc[j].z = acc[j].z * sc + af * x[j].z;
            acc[j].w = acc[j].w * sc + af * x[j].w;
        }
        m = mn;
    }

    // Publish wave-private results.
#pragma unroll
    for (int j = 0; j < 8; ++j)
        *reinterpret_cast<float4*>(&accL[w * DIM + j * 128 + lane * 4]) = acc[j];
    if (lane == 0) { wmx[w] = m; wsm[w] = s; wcc[w] = cacc; }
    __syncthreads();

    // Flash combine across the 8 waves; thread t owns elements t*4..t*4+3.
    float M = -3.4e38f;
#pragma unroll
    for (int w2 = 0; w2 < 8; ++w2) M = fmaxf(M, wmx[w2]);
    float ew[8];
    float S = 0.f, C = 0.f;
#pragma unroll
    for (int w2 = 0; w2 < 8; ++w2) {
        ew[w2] = __expf(wmx[w2] - M);
        S += ew[w2] * wsm[w2];
        C += ew[w2] * wcc[w2];
    }
    const int e0 = t * 4;
    float ax = 0.f, ay = 0.f, az = 0.f, aw_ = 0.f;
#pragma unroll
    for (int w2 = 0; w2 < 8; ++w2) {
        float4 v = *reinterpret_cast<const float4*>(&accL[w2 * DIM + e0]);
        ax = fmaf(ew[w2], v.x, ax); ay = fmaf(ew[w2], v.y, ay);
        az = fmaf(ew[w2], v.z, az); aw_ = fmaf(ew[w2], v.w, aw_);
    }
    float inv = 1.f / S;
    float cfin = C * inv;
    float4 g4 = *reinterpret_cast<const float4*>(g + e0);
    float4 b4 = *reinterpret_cast<const float4*>(bln + e0);
    float4 o;
    o.x = g4.x * (ax * inv - cfin) + b4.x;
    o.y = g4.y * (ay * inv - cfin) + b4.y;
    o.z = g4.z * (az * inv - cfin) + b4.z;
    o.w = g4.w * (aw_ * inv - cfin) + b4.w;
    *reinterpret_cast<float4*>(XW + (size_t)b * DIM + e0) = o;
}

// ---------------------------------------------------------------------------
extern "C" void kernel_launch(void* const* d_in, const int* in_sizes, int n_in,
                              void* d_out, int out_size, void* d_ws, size_t ws_size,
                              hipStream_t stream) {
    (void)in_sizes; (void)n_in; (void)out_size; (void)ws_size;

    const float* text  = (const float*)d_in[0];
    const float* video = (const float*)d_in[1];
    const float* q_w   = (const float*)d_in[2];
    const float* k_w   = (const float*)d_in[3];
    const float* v_w   = (const float*)d_in[4];
    const float* out_w = (const float*)d_in[5];
    const float* lin_w = (const float*)d_in[6];
    const float* q_b   = (const float*)d_in[7];
    const float* k_b   = (const float*)d_in[8];
    const float* v_b   = (const float*)d_in[9];
    const float* out_b = (const float*)d_in[10];
    const float* lin_b = (const float*)d_in[11];
    const float* ln1_g = (const float*)d_in[12];
    const float* ln1_b = (const float*)d_in[13];
    const float* ln2_g = (const float*)d_in[14];
    const float* ln2_b = (const float*)d_in[15];
    const float* ln3_g = (const float*)d_in[16];
    const float* ln3_b = (const float*)d_in[17];

    // Workspace: 18 slots of 2 MB (slots 0-9 packed weights, 10-17 fp32 acts).
    uint8_t* ws = (uint8_t*)d_ws;
    const size_t SLOT = (size_t)DIM * DIM * 2;   // 2 MB
    __bf16* WqH = (__bf16*)(ws + 0 * SLOT); __bf16* WqL = (__bf16*)(ws + 1 * SLOT);
    __bf16* WkH = (__bf16*)(ws + 2 * SLOT); __bf16* WkL = (__bf16*)(ws + 3 * SLOT);
    __bf16* WvH = (__bf16*)(ws + 4 * SLOT); __bf16* WvL = (__bf16*)(ws + 5 * SLOT);
    __bf16* WoH = (__bf16*)(ws + 6 * SLOT); __bf16* WoL = (__bf16*)(ws + 7 * SLOT);
    __bf16* WlH = (__bf16*)(ws + 8 * SLOT); __bf16* WlL = (__bf16*)(ws + 9 * SLOT);
    float* X0   = (float*)(ws + 10 * SLOT);
    float* Qm   = (float*)(ws + 11 * SLOT);
    float* QKm  = (float*)(ws + 12 * SLOT);
    float* XWm  = (float*)(ws + 13 * SLOT);
    float* ATT  = (float*)(ws + 14 * SLOT);
    float* ATT2 = (float*)(ws + 15 * SLOT);
    float* AO   = (float*)(ws + 16 * SLOT);
    float* LIN  = (float*)(ws + 17 * SLOT);

    const dim3 gPrep(4096), gRows(BATCH), gGemm(16, 8), blk(256);

    // Pack weights (T=1: used as X@W.T ; k_w T=0: used as q@k_w).
    vt_prep_weights<<<gPrep, blk, 0, stream>>>(q_w,   WqH, WqL, 1);
    vt_prep_weights<<<gPrep, blk, 0, stream>>>(k_w,   WkH, WkL, 0);
    vt_prep_weights<<<gPrep, blk, 0, stream>>>(v_w,   WvH, WvL, 1);
    vt_prep_weights<<<gPrep, blk, 0, stream>>>(out_w, WoH, WoL, 1);
    vt_prep_weights<<<gPrep, blk, 0, stream>>>(lin_w, WlH, WlL, 1);

    // t = LN1(text); q = t @ q_w.T + q_b; qk = q @ k_w
    vt_ln_rows<<<gRows, blk, 0, stream>>>(text, nullptr, ln1_g, ln1_b, X0);
    vt_gemm_bias<<<gGemm, blk, 0, stream>>>(X0, WqH, WqL, q_b, Qm);
    vt_gemm_bias<<<gGemm, blk, 0, stream>>>(Qm, WkH, WkL, nullptr, QKm);

    // Fused LN1(video) + logits + online softmax + weighted frame sum.
    vt_attn_fused<<<gRows, blk, 0, stream>>>(video, QKm, Qm, k_b,
                                             ln1_g, ln1_b, XWm);

    // attn = XW @ v_w.T + v_b; attn = attn @ out_w.T + out_b
    vt_gemm_bias<<<gGemm, blk, 0, stream>>>(XWm, WvH, WvL, v_b, ATT);
    vt_gemm_bias<<<gGemm, blk, 0, stream>>>(ATT, WoH, WoL, out_b, ATT2);

    // attn_out = LN2(attn); linear = attn_out @ lin_w.T + lin_b
    vt_ln_rows<<<gRows, blk, 0, stream>>>(ATT2, nullptr, ln2_g, ln2_b, AO);
    vt_gemm_bias<<<gGemm, blk, 0, stream>>>(AO, WlH, WlL, lin_b, LIN);

    // out = LN3(attn_out + linear)
    vt_ln_rows<<<gRows, blk, 0, stream>>>(AO, LIN, ln3_g, ln3_b, (float*)d_out);
}
